// ConformHopfieldBatchSameEnc_30794915512940
// MI455X (gfx1250) — matrix operand order repaired and checked by reading, not compiled, not run on hardware
//
#include <hip/hip_runtime.h>
#include <hip/hip_bf16.h>

// ---------------------------------------------------------------------------
// ConformHopfieldBatchSameEnc on MI455X (gfx1250, wave32, WMMA bf16 16x16x32)
// dims: B=16 M=4 S=512 DIN=64 E=4 DOUT=128 H=4 DH=128
// Weights are pre-swizzled into WMMA B-fragment layout so the GEMM inner loop
// is pure vector loads + v_wmma.
// ---------------------------------------------------------------------------

typedef __attribute__((ext_vector_type(16))) __bf16 v16bf;
typedef __attribute__((ext_vector_type(8)))  __bf16 v8bf;
typedef __attribute__((ext_vector_type(8)))  float  v8f;

#define BB    16
#define MM    4
#define SS    512
#define DIN_  64
#define EE    4
#define DOUT_ 128
#define HH    4
#define DH_   128
#define ROWS1 (BB*MM*SS)      /* 32768 rows per input tensor */
#define ROWS2 (2*ROWS1)       /* 65536: X_true rows then X_sim rows */
#define ENCH  640             /* 600 padded to 20*32 */
#define HDH   256             /* 200 padded to 8*32  */
#define QKN   (HH*DH_)        /* 512 */

// ---------------------------------------------------------------------------
// Pack weights [models][K][N] f32 -> WMMA B-fragment layout bf16:
//   dst[((kt*ntiles + nt)*32 + lane)*16 + i] = W[kt*32 + (lane>>4)*16 + i][nt*16 + (lane&15)]
// (k order used here, same as the A loads: lanes 0-15 hold K 0..15, lanes 16-31 hold K 16..31, 2 K / VGPR)
// Zero-padded to (Kp, Np).
// ---------------------------------------------------------------------------
__global__ void pack_w_frag(const float* __restrict__ src, __bf16* __restrict__ dst,
                            int K, int N, int Kp, int Np, int models) {
  long idx = (long)blockIdx.x * blockDim.x + threadIdx.x;
  long per = (long)Kp * Np;
  if (idx >= per * models) return;
  int  mdl  = (int)(idx / per);
  long r    = idx % per;
  int  i    = (int)(r & 15);
  int  lane = (int)((r >> 4) & 31);
  long tile = r >> 9;
  int  ntiles = Np >> 4;
  int  kt = (int)(tile / ntiles), nt = (int)(tile % ntiles);
  int  k  = kt * 32 + ((lane >> 4) << 4) + i;
  int  n  = nt * 16 + (lane & 15);
  float v = (k < K && n < N) ? src[((long)mdl * K + k) * N + n] : 0.0f;
  dst[idx] = (__bf16)v;
}

__global__ void pack_b_f32(const float* __restrict__ src, float* __restrict__ dst,
                           int N, int Np, int models) {
  int idx = blockIdx.x * blockDim.x + threadIdx.x;
  if (idx >= Np * models) return;
  int mdl = idx / Np, np = idx % Np;
  dst[idx] = (np < N) ? src[mdl * N + np] : 0.0f;
}

__global__ void cvt_inputs(const float* __restrict__ Xt, const float* __restrict__ Xs,
                           __bf16* __restrict__ dst, long nPer) {
  long idx = (long)blockIdx.x * blockDim.x + threadIdx.x;
  if (idx >= 2 * nPer) return;
  float v = (idx < nPer) ? Xt[idx] : Xs[idx - nPer];
  dst[idx] = (__bf16)v;
}

// ---------------------------------------------------------------------------
// bf16 WMMA GEMM with fragment-layout weights.
//  C[row, n] = act( A[row, :K] * W[:K, n] + bias[n] )
//  - block = 128 thr (4 waves); wave w owns rows [blockIdx.y*64 + w*16, +16)
//    and 64 cols [blockIdx.x*64, +64) -> 4 accumulators / wave.
//  - grid = (N/64, ROWS2/64)
//  - per-model weights: m = (row/512)%4; rows >= halfRows use Wsel (Wq|Wk).
// ---------------------------------------------------------------------------
__global__ void __launch_bounds__(128)
gemm_bf16_wmma(const __bf16* __restrict__ A,
               const __bf16* __restrict__ W0,
               const __bf16* __restrict__ Wsel,
               const float*  __restrict__ bias,
               __bf16* __restrict__ C,
               int K, int N,
               long wstride, long bstride,
               int halfRows, int relu) {
  const int lane = threadIdx.x & 31;
  const int wave = threadIdx.x >> 5;
  const int row0 = (blockIdx.y * 4 + wave) * 16;
  const int n0   = blockIdx.x * 64;
  const int m    = (row0 >> 9) & (MM - 1);
  const int ntiles = N >> 4;
  const int nt0    = n0 >> 4;

  const __bf16* W = ((Wsel != nullptr) && (row0 >= halfRows)) ? Wsel : W0;
  W += (long)m * wstride;
  const float* bp = bias ? (bias + (long)m * bstride) : nullptr;

  const int arow  = row0 + (lane & 15);
  const int ahalf = (lane >> 4) * 16;     // A frag: half selects K chunks {0..15} vs {16..31}

  v8f c0 = {}, c1 = {}, c2 = {}, c3 = {};
  const __bf16* Ar = A + (long)arow * K;
  for (int k = 0; k < K; k += 32) {
    union { v16bf v; v8bf h[2]; } a;
    a.h[0] = *(const v8bf*)(Ar + k + ahalf);
    a.h[1] = *(const v8bf*)(Ar + k + 8 + ahalf);
    __builtin_prefetch(Ar + k + 256, 0, 1);   // streaming A (HBM-resident)
    const __bf16* Bt = W + ((long)(k >> 5) * ntiles + nt0) * 512 + lane * 16;
    v16bf b0 = *(const v16bf*)(Bt);
    v16bf b1 = *(const v16bf*)(Bt + 512);
    v16bf b2 = *(const v16bf*)(Bt + 1024);
    v16bf b3 = *(const v16bf*)(Bt + 1536);
    c0 = __builtin_amdgcn_wmma_f32_16x16x32_bf16(false, a.v, false, b0, (short)0, c0, false, false);
    c1 = __builtin_amdgcn_wmma_f32_16x16x32_bf16(false, a.v, false, b1, (short)0, c1, false, false);
    c2 = __builtin_amdgcn_wmma_f32_16x16x32_bf16(false, a.v, false, b2, (short)0, c2, false, false);
    c3 = __builtin_amdgcn_wmma_f32_16x16x32_bf16(false, a.v, false, b3, (short)0, c3, false, false);
  }
  // D layout: VGPR v -> row (v + 8*half), col = lane%16
  const int rbase = row0 + 8 * (lane >> 4);
#pragma unroll
  for (int t = 0; t < 4; ++t) {
    const v8f* cp = (t == 0) ? &c0 : (t == 1) ? &c1 : (t == 2) ? &c2 : &c3;
    const int ncol = n0 + t * 16 + (lane & 15);
    const float bv = bp ? bp[ncol] : 0.0f;
#pragma unroll
    for (int v = 0; v < 8; ++v) {
      float x = (*cp)[v] + bv;
      if (relu) x = fmaxf(x, 0.0f);
      C[(long)(rbase + v) * N + ncol] = (__bf16)x;
    }
  }
}

// ---------------------------------------------------------------------------
// LayerNorm over 128 cols, per-model affine; rows<32768 -> lnq else lnk.
// block = 256 thr = 8 waves, one row per wave.
// ---------------------------------------------------------------------------
__global__ void __launch_bounds__(256)
ln_kernel(const __bf16* __restrict__ X,
          const float* __restrict__ gq, const float* __restrict__ bq,
          const float* __restrict__ gk, const float* __restrict__ bk,
          __bf16* __restrict__ Y) {
  const int  lane = threadIdx.x & 31;
  const long row  = (long)blockIdx.x * 8 + (threadIdx.x >> 5);
  const __bf16* xr = X + row * DOUT_;
  float x[4];
#pragma unroll
  for (int i = 0; i < 4; ++i) x[i] = (float)xr[lane + 32 * i];
  float s = x[0] + x[1] + x[2] + x[3];
  for (int o = 16; o; o >>= 1) s += __shfl_xor(s, o);
  const float mu = s * (1.0f / DOUT_);
  float vs = 0.0f;
#pragma unroll
  for (int i = 0; i < 4; ++i) { float d = x[i] - mu; vs += d * d; }
  for (int o = 16; o; o >>= 1) vs += __shfl_xor(vs, o);
  const float inv = rsqrtf(vs * (1.0f / DOUT_) + 1e-5f);
  const int   m   = (int)((row >> 9) & (MM - 1));
  const float* g = (row < ROWS1 ? gq : gk) + m * DOUT_;
  const float* b = (row < ROWS1 ? bq : bk) + m * DOUT_;
  __bf16* yr = Y + row * DOUT_;
#pragma unroll
  for (int i = 0; i < 4; ++i) {
    int cidx = lane + 32 * i;
    yr[cidx] = (__bf16)((x[i] - mu) * inv * g[cidx] + b[cidx]);
  }
}

// ---------------------------------------------------------------------------
// Fused Hopfield attention + loss partials.
// One block (128 thr, 4 waves) per (b, m, h, 16-query tile):
//   WMMA scores 16x512 -> LDS -> masked softmax -> dot with v -> (out-v)^2.
// Q fragments hoisted (4 frags reused across 8 key tiles / wave).
// ---------------------------------------------------------------------------
#define SC_STRIDE 520
__global__ void __launch_bounds__(128)
attn_fused(const __bf16* __restrict__ QK,      // [65536,512]; first half Q, second half K
           const float* __restrict__ errors,   // [B,M,S,E] f32
           const int* __restrict__ whichp,
           float* __restrict__ partials) {
  __shared__ float sc[16][SC_STRIDE];
  __shared__ float wsum[4];
  const int lane = threadIdx.x & 31;
  const int wave = threadIdx.x >> 5;
  int bid = blockIdx.x;
  const int qt = bid & 31; bid >>= 5;
  const int h  = bid & 3;  bid >>= 2;
  const int m  = bid & 3;  bid >>= 2;
  const int b  = bid;
  const long seqBase = ((long)b * MM + m) * SS;
  const __bf16* Qb = QK + (seqBase + qt * 16) * QKN + h * DH_;
  const __bf16* Kb = QK + ((long)ROWS1 + seqBase) * QKN + h * DH_;

  const int arow  = lane & 15;
  const int ahalf = (lane >> 4) * 16;
  const int bcol  = lane & 15;
  const int bkb   = (lane >> 4) * 16;
  const float scale = 0.08838834764831845f;    // 1/sqrt(DOUT)

  // hoisted Q fragments: kk = 0,32,64,96
  union { v16bf v; v8bf h2[2]; } a[4];
  const __bf16* Qr = Qb + (long)arow * QKN;
#pragma unroll
  for (int kk4 = 0; kk4 < 4; ++kk4) {
    a[kk4].h2[0] = *(const v8bf*)(Qr + kk4 * 32 + ahalf);
    a[kk4].h2[1] = *(const v8bf*)(Qr + kk4 * 32 + 8 + ahalf);
  }

  for (int kt = wave * 8; kt < wave * 8 + 8; ++kt) {
    v8f c = {};
    const __bf16* Kr = Kb + (long)(kt * 16 + bcol) * QKN + bkb;
#pragma unroll
    for (int kk4 = 0; kk4 < 4; ++kk4) {
      v16bf bb = *(const v16bf*)(Kr + kk4 * 32);
      c = __builtin_amdgcn_wmma_f32_16x16x32_bf16(false, a[kk4].v, false, bb,
                                                  (short)0, c, false, false);
    }
#pragma unroll
    for (int v = 0; v < 8; ++v)
      sc[v + 8 * (lane >> 4)][kt * 16 + (lane & 15)] = c[v] * scale;
  }
  __syncthreads();

  const int which = *whichp;
  const float* ev = errors + seqBase * EE + which;   // stride EE per key
  float acc = 0.0f;
  for (int r = wave * 4; r < wave * 4 + 4; ++r) {
    const int qg = qt * 16 + r;                      // masked diagonal key
    float mx = -3.4e38f;
    for (int j = lane; j < SS; j += 32)
      if (j != qg) mx = fmaxf(mx, sc[r][j]);
    for (int o = 16; o; o >>= 1) mx = fmaxf(mx, __shfl_xor(mx, o));
    float se = 0.0f, sv = 0.0f;
    for (int j = lane; j < SS; j += 32) {
      if (j == qg) continue;
      float e = __expf(sc[r][j] - mx);
      se += e;
      sv += e * ev[(long)j * EE];
    }
    for (int o = 16; o; o >>= 1) { se += __shfl_xor(se, o); sv += __shfl_xor(sv, o); }
    const float out = sv / se;
    const float d   = out - ev[(long)qg * EE];
    acc += d * d;
  }
  if (lane == 0) wsum[wave] = acc;
  __syncthreads();
  if (threadIdx.x == 0)
    partials[blockIdx.x] = wsum[0] + wsum[1] + wsum[2] + wsum[3];
}

// Deterministic per-model reduction of partials -> losses[M]
__global__ void __launch_bounds__(128)
reduce_losses(const float* __restrict__ partials, float* __restrict__ out) {
  const int lane = threadIdx.x & 31;
  const int m    = threadIdx.x >> 5;    // 4 waves, one per model
  float s = 0.0f;
  for (int t = lane; t < BB * HH * 32; t += 32) {   // 2048 partials per model
    int qt = t & 31, h = (t >> 5) & 3, b = t >> 7;
    s += partials[(((b * MM + m) * HH + h) << 5) + qt];
  }
  for (int o = 16; o; o >>= 1) s += __shfl_xor(s, o);
  if (lane == 0) out[m] = s * (1.0f / (float)(BB * SS * HH));
}

// ---------------------------------------------------------------------------
// Launch
// ---------------------------------------------------------------------------
static inline size_t bump(size_t& o, size_t bytes) {
  size_t r = o; o = (o + bytes + 255) & ~(size_t)255; return r;
}

extern "C" void kernel_launch(void* const* d_in, const int* in_sizes, int n_in,
                              void* d_out, int out_size, void* d_ws, size_t ws_size,
                              hipStream_t stream) {
  (void)in_sizes; (void)n_in; (void)out_size; (void)ws_size;
  const float* X_true = (const float*)d_in[0];
  const float* X_sim  = (const float*)d_in[1];
  const float* errors = (const float*)d_in[2];
  const float* eW1 = (const float*)d_in[3];  const float* eb1 = (const float*)d_in[4];
  const float* eW2 = (const float*)d_in[5];  const float* eb2 = (const float*)d_in[6];
  const float* eW3 = (const float*)d_in[7];  const float* eb3 = (const float*)d_in[8];
  const float* hW1 = (const float*)d_in[9];  const float* hb1 = (const float*)d_in[10];
  const float* hW2 = (const float*)d_in[11]; const float* hb2 = (const float*)d_in[12];
  const float* hW3 = (const float*)d_in[13]; const float* hb3 = (const float*)d_in[14];
  const float* lnqg = (const float*)d_in[15]; const float* lnqb = (const float*)d_in[16];
  const float* lnkg = (const float*)d_in[17]; const float* lnkb = (const float*)d_in[18];
  const float* Wq = (const float*)d_in[19];  const float* Wk = (const float*)d_in[20];
  const int* which = (const int*)d_in[21];
  float* out = (float*)d_out;

  char* ws = (char*)d_ws;
  size_t o = 0;
  // activations (bf16)
  __bf16* XBF  = (__bf16*)(ws + bump(o, (size_t)ROWS2 * DIN_  * 2));
  __bf16* BUFA = (__bf16*)(ws + bump(o, (size_t)ROWS2 * ENCH  * 2)); // H1 / headH1 / QK
  __bf16* BUFB = (__bf16*)(ws + bump(o, (size_t)ROWS2 * ENCH  * 2)); // H2 / headH2
  __bf16* ENC  = (__bf16*)(ws + bump(o, (size_t)ROWS2 * DOUT_ * 2));
  __bf16* HOUT = (__bf16*)(ws + bump(o, (size_t)ROWS2 * DOUT_ * 2));
  __bf16* QKLN = (__bf16*)(ws + bump(o, (size_t)ROWS2 * DOUT_ * 2));
  // packed bf16 weights, WMMA B-fragment layout (zero-padded)
  __bf16* W1B  = (__bf16*)(ws + bump(o, (size_t)DIN_ * ENCH * 2));
  __bf16* W2B  = (__bf16*)(ws + bump(o, (size_t)ENCH * ENCH * 2));
  __bf16* W3B  = (__bf16*)(ws + bump(o, (size_t)ENCH * DOUT_ * 2));
  __bf16* HW1B = (__bf16*)(ws + bump(o, (size_t)MM * DOUT_ * HDH * 2));
  __bf16* HW2B = (__bf16*)(ws + bump(o, (size_t)MM * HDH * HDH * 2));
  __bf16* HW3B = (__bf16*)(ws + bump(o, (size_t)MM * HDH * DOUT_ * 2));
  __bf16* WQB  = (__bf16*)(ws + bump(o, (size_t)MM * DOUT_ * QKN * 2));
  __bf16* WKB  = (__bf16*)(ws + bump(o, (size_t)MM * DOUT_ * QKN * 2));
  // padded f32 biases
  float* B1P  = (float*)(ws + bump(o, (size_t)ENCH * 4));
  float* B2P  = (float*)(ws + bump(o, (size_t)ENCH * 4));
  float* B3P  = (float*)(ws + bump(o, (size_t)DOUT_ * 4));
  float* HB1P = (float*)(ws + bump(o, (size_t)MM * HDH * 4));
  float* HB2P = (float*)(ws + bump(o, (size_t)MM * HDH * 4));
  float* HB3P = (float*)(ws + bump(o, (size_t)MM * DOUT_ * 4));
  float* PART = (float*)(ws + bump(o, (size_t)(BB * MM * HH * 32) * 4));

  const int T = 256;
  auto gsz = [](long n, int t) { return (unsigned)((n + t - 1) / t); };

  // ---- pack weights / biases, convert inputs ------------------------------
  pack_w_frag<<<gsz((long)DIN_*ENCH,T),T,0,stream>>>(eW1, W1B, DIN_, 600, DIN_, ENCH, 1);
  pack_w_frag<<<gsz((long)ENCH*ENCH,T),T,0,stream>>>(eW2, W2B, 600, 600, ENCH, ENCH, 1);
  pack_w_frag<<<gsz((long)ENCH*DOUT_,T),T,0,stream>>>(eW3, W3B, 600, DOUT_, ENCH, DOUT_, 1);
  pack_w_frag<<<gsz((long)MM*DOUT_*HDH,T),T,0,stream>>>(hW1, HW1B, DOUT_, 200, DOUT_, HDH, MM);
  pack_w_frag<<<gsz((long)MM*HDH*HDH,T),T,0,stream>>>(hW2, HW2B, 200, 200, HDH, HDH, MM);
  pack_w_frag<<<gsz((long)MM*HDH*DOUT_,T),T,0,stream>>>(hW3, HW3B, 200, DOUT_, HDH, DOUT_, MM);
  pack_w_frag<<<gsz((long)MM*DOUT_*QKN,T),T,0,stream>>>(Wq, WQB, DOUT_, QKN, DOUT_, QKN, MM);
  pack_w_frag<<<gsz((long)MM*DOUT_*QKN,T),T,0,stream>>>(Wk, WKB, DOUT_, QKN, DOUT_, QKN, MM);
  pack_b_f32<<<gsz(ENCH,T),T,0,stream>>>(eb1, B1P, 600, ENCH, 1);
  pack_b_f32<<<gsz(ENCH,T),T,0,stream>>>(eb2, B2P, 600, ENCH, 1);
  pack_b_f32<<<gsz(DOUT_,T),T,0,stream>>>(eb3, B3P, DOUT_, DOUT_, 1);
  pack_b_f32<<<gsz(MM*HDH,T),T,0,stream>>>(hb1, HB1P, 200, HDH, MM);
  pack_b_f32<<<gsz(MM*HDH,T),T,0,stream>>>(hb2, HB2P, 200, HDH, MM);
  pack_b_f32<<<gsz(MM*DOUT_,T),T,0,stream>>>(hb3, HB3P, DOUT_, DOUT_, MM);
  cvt_inputs<<<gsz((long)ROWS2*DIN_,T),T,0,stream>>>(X_true, X_sim, XBF, (long)ROWS1*DIN_);

  // ---- shared encoder (rows = X_true ++ X_sim) ----------------------------
  dim3 blk(128);
  gemm_bf16_wmma<<<dim3(ENCH/64, ROWS2/64), blk, 0, stream>>>(
      XBF, W1B, nullptr, B1P, BUFA, DIN_, ENCH, 0, 0, ROWS2, 1);
  gemm_bf16_wmma<<<dim3(ENCH/64, ROWS2/64), blk, 0, stream>>>(
      BUFA, W2B, nullptr, B2P, BUFB, ENCH, ENCH, 0, 0, ROWS2, 1);
  gemm_bf16_wmma<<<dim3(DOUT_/64, ROWS2/64), blk, 0, stream>>>(
      BUFB, W3B, nullptr, B3P, ENC, ENCH, DOUT_, 0, 0, ROWS2, 0);

  // ---- per-model heads ----------------------------------------------------
  gemm_bf16_wmma<<<dim3(HDH/64, ROWS2/64), blk, 0, stream>>>(
      ENC, HW1B, nullptr, HB1P, BUFA, DOUT_, HDH, (long)DOUT_*HDH, HDH, ROWS2, 1);
  gemm_bf16_wmma<<<dim3(HDH/64, ROWS2/64), blk, 0, stream>>>(
      BUFA, HW2B, nullptr, HB2P, BUFB, HDH, HDH, (long)HDH*HDH, HDH, ROWS2, 1);
  gemm_bf16_wmma<<<dim3(DOUT_/64, ROWS2/64), blk, 0, stream>>>(
      BUFB, HW3B, nullptr, HB3P, HOUT, HDH, DOUT_, (long)HDH*DOUT_, DOUT_, ROWS2, 0);

  // ---- hopfield layernorms ------------------------------------------------
  ln_kernel<<<ROWS2/8, 256, 0, stream>>>(HOUT, lnqg, lnqb, lnkg, lnkb, QKLN);

  // ---- Q/K projections (Wq first half of rows, Wk second half) ------------
  gemm_bf16_wmma<<<dim3(QKN/64, ROWS2/64), blk, 0, stream>>>(
      QKLN, WQB, WKB, nullptr, BUFA /* QK */, DOUT_, QKN,
      (long)DOUT_*QKN, 0, ROWS1, 0);

  // ---- fused attention + loss partials, deterministic reduce --------------
  attn_fused<<<BB*MM*HH*32, 128, 0, stream>>>(BUFA, errors, which, PART);
  reduce_losses<<<1, 128, 0, stream>>>(PART, out);
}
